// EPMoE_17136919511769
// MI455X (gfx1250) — matrix operand order repaired and checked by reading, not compiled, not run on hardware
//
#include <hip/hip_runtime.h>
#include <math.h>

// Problem constants (match reference)
#define T_TOK 1024
#define KTOP  2
#define NEXP  8
#define HDIM  1024
#define FDIM  1024
#define TK    (T_TOK * KTOP)        // 2048 expanded rows
#define TILE_M 32
#define MAX_TILES (TK / TILE_M)     // 64
#define RSU 516                     // LDS row stride in u32 (512 pairs + 4 pad)
#define SMEM_BYTES ((2 * TILE_M * RSU) * 4 + TILE_M * 4 + TILE_M * 4)
#define MATELEMS ((size_t)HDIM * FDIM)          // 1M elements per expert matrix

typedef __attribute__((ext_vector_type(16))) __bf16 v16bf;
typedef __attribute__((ext_vector_type(8)))  float  v8f;

union Frag { v16bf v; unsigned u[8]; uint4 q[2]; };

__device__ __forceinline__ unsigned pack2(float a, float b) {
  union { __bf16 h[2]; unsigned u; } p;
  p.h[0] = (__bf16)a;
  p.h[1] = (__bf16)b;
  return p.u;
}

__device__ __forceinline__ unsigned short bf16bits(float f) {
  union { __bf16 h; unsigned short u; } p;
  p.h = (__bf16)f;
  return p.u;
}

__device__ __forceinline__ float silu_fast(float g) {
  return g * __builtin_amdgcn_rcpf(1.f + __expf(-g));  // g * sigmoid(g)
}

__global__ void zero_kernel(float4* __restrict__ out4, int* __restrict__ counts) {
  int idx = blockIdx.x * blockDim.x + threadIdx.x;
  out4[idx] = make_float4(0.f, 0.f, 0.f, 0.f);
  if (blockIdx.x == 0 && threadIdx.x < NEXP) counts[threadIdx.x] = 0;
}

__global__ void route_kernel(const int* __restrict__ ids, int* __restrict__ counts,
                             int* __restrict__ list) {
  int i = blockIdx.x * blockDim.x + threadIdx.x;
  if (i < TK) {
    int e = ids[i];
    int pos = atomicAdd(&counts[e], 1);
    list[e * TK + pos] = i;
  }
}

// Transpose + convert one 1024x1024 f32 matrix per (mat, expert) into bf16 [col][row].
__global__ __launch_bounds__(256) void wtrans_kernel(
    const float* __restrict__ wi0, const float* __restrict__ wi1,
    const float* __restrict__ wo,
    unsigned short* __restrict__ t0, unsigned short* __restrict__ t1,
    unsigned short* __restrict__ t2)
{
  __shared__ unsigned short tile[32][33];
  const int mat = blockIdx.z / NEXP;
  const int e   = blockIdx.z % NEXP;
  const float* src = (mat == 0 ? wi0 : mat == 1 ? wi1 : wo) + (size_t)e * MATELEMS;
  unsigned short* dst = (mat == 0 ? t0 : mat == 1 ? t1 : t2) + (size_t)e * MATELEMS;
  const int bx = blockIdx.x * 32;   // src col
  const int by = blockIdx.y * 32;   // src row
  const int lx = threadIdx.x;       // 0..31
  const int ly = threadIdx.y;       // 0..7
  #pragma unroll
  for (int i = 0; i < 32; i += 8)
    tile[ly + i][lx] = bf16bits(src[(size_t)(by + ly + i) * 1024 + bx + lx]);
  __syncthreads();
  #pragma unroll
  for (int i = 0; i < 32; i += 8)
    dst[(size_t)(bx + ly + i) * 1024 + by + lx] = tile[lx][ly + i];
}

// BF16W=1: weights are bf16, transposed [out_col][K] -> B frag = 2x global_load_b128.
// BF16W=0: weights are f32 row-major -> strided gather + in-flight convert (fallback).
template <bool BF16W>
__global__ __launch_bounds__(256) void moe_kernel(
    const float* __restrict__ x, const float* __restrict__ tw,
    const void* __restrict__ w0p, const void* __restrict__ w1p,
    const void* __restrict__ wop,
    const int* __restrict__ counts, const int* __restrict__ list,
    float* __restrict__ out)
{
  extern __shared__ unsigned smem[];
  unsigned* ldsX = smem;                       // [32][516] packed bf16 pairs of X tile
  unsigned* ldsI = smem + TILE_M * RSU;        // [32][516] packed bf16 pairs of inter
  int*   sTok = (int*)(smem + 2 * TILE_M * RSU);
  float* sW   = (float*)(sTok + TILE_M);

  const int bid  = blockIdx.x;
  const int e    = bid / MAX_TILES;
  const int tile = bid % MAX_TILES;
  const int ne   = counts[e];
  if (tile * TILE_M >= ne) return;             // block-uniform early exit

  const int tid  = threadIdx.x;
  const int lane = tid & 31;
  const int wv   = tid >> 5;                   // wave 0..7
  const int lh   = lane >> 4;                  // lane half
  const int m    = lane & 15;

  if (tid < TILE_M) {
    int r = tile * TILE_M + tid;
    int tok = 0; float w = 0.f;
    if (r < ne) {
      int i = list[e * TK + r];
      tok = i >> 1;                            // K == 2
      w = tw[i];
    }
    sTok[tid] = tok; sW[tid] = w;
  }
  __syncthreads();

  // gather X tile -> bf16 LDS (32 x 1024, packed pairs)
  for (int p = tid; p < TILE_M * (HDIM / 2); p += 256) {
    int row = p >> 9;
    int kp  = p & 511;
    const float* src = x + (size_t)sTok[row] * HDIM + kp * 2;
    ldsX[row * RSU + kp] = pack2(src[0], src[1]);
  }
  __syncthreads();

  const float* w0f = (const float*)w0p + (size_t)e * MATELEMS;
  const float* w1f = (const float*)w1p + (size_t)e * MATELEMS;
  const float* wof = (const float*)wop + (size_t)e * MATELEMS;
  const unsigned short* w0t = (const unsigned short*)w0p + (size_t)e * MATELEMS;
  const unsigned short* w1t = (const unsigned short*)w1p + (size_t)e * MATELEMS;
  const unsigned short* wot = (const unsigned short*)wop + (size_t)e * MATELEMS;
  __bf16* ldsIh = (__bf16*)ldsI;

  // ================= Stage 1: inter = silu(X*wi0) * (X*wi1) =================
  for (int c = 0; c < 8; ++c) {
    const int n0 = c * 128 + wv * 16;
    v8f accG0 = {}, accG1 = {}, accU0 = {}, accU1 = {};
    for (int k0 = 0; k0 < HDIM; k0 += 32) {
      const int kp0 = k0 >> 1;
      Frag a0, a1, b0, b1;
      {
        const uint4* r0 = (const uint4*)(ldsX + (m)      * RSU + kp0 + 8 * lh);
        const uint4* r1 = (const uint4*)(ldsX + (16 + m) * RSU + kp0 + 8 * lh);
        a0.q[0] = r0[0]; a0.q[1] = r0[1];
        a1.q[0] = r1[0]; a1.q[1] = r1[1];
      }
      if constexpr (BF16W) {
        // transposed bf16: 16 contiguous K values per lane = 32B = 2x b128
        const uint4* p0 = (const uint4*)(w0t + (size_t)(n0 + m) * HDIM + k0 + 16 * lh);
        const uint4* p1 = (const uint4*)(w1t + (size_t)(n0 + m) * HDIM + k0 + 16 * lh);
        b0.q[0] = p0[0]; b0.q[1] = p0[1];
        b1.q[0] = p1[0]; b1.q[1] = p1[1];
      } else {
        #pragma unroll
        for (int j = 0; j < 8; ++j) {
          int kk = k0 + 16 * lh + 2 * j;
          const float* p0 = w0f + (size_t)kk * FDIM + n0 + m;
          const float* p1 = w1f + (size_t)kk * FDIM + n0 + m;
          b0.u[j] = pack2(p0[0], p0[FDIM]);
          b1.u[j] = pack2(p1[0], p1[FDIM]);
        }
      }
      accG0 = __builtin_amdgcn_wmma_f32_16x16x32_bf16(false, a0.v, false, b0.v, (short)0, accG0, false, false);
      accG1 = __builtin_amdgcn_wmma_f32_16x16x32_bf16(false, a1.v, false, b0.v, (short)0, accG1, false, false);
      accU0 = __builtin_amdgcn_wmma_f32_16x16x32_bf16(false, a0.v, false, b1.v, (short)0, accU0, false, false);
      accU1 = __builtin_amdgcn_wmma_f32_16x16x32_bf16(false, a1.v, false, b1.v, (short)0, accU1, false, false);
    }
    #pragma unroll
    for (int r = 0; r < 8; ++r) {
      {
        float s = silu_fast(accG0[r]) * accU0[r];
        ldsIh[(r + 8 * lh) * (2 * RSU) + n0 + m] = (__bf16)s;
      }
      {
        float s = silu_fast(accG1[r]) * accU1[r];
        ldsIh[(16 + r + 8 * lh) * (2 * RSU) + n0 + m] = (__bf16)s;
      }
    }
  }
  __syncthreads();

  // ================= Stage 2: y = inter @ wo, weighted atomic combine =================
  for (int g = 0; g < 8; ++g) {
    const int h0 = g * 128 + wv * 16;
    v8f acc0 = {}, acc1 = {};
    for (int k0 = 0; k0 < FDIM; k0 += 32) {
      const int kp0 = k0 >> 1;
      Frag a0, a1, b;
      {
        const uint4* r0 = (const uint4*)(ldsI + (m)      * RSU + kp0 + 8 * lh);
        const uint4* r1 = (const uint4*)(ldsI + (16 + m) * RSU + kp0 + 8 * lh);
        a0.q[0] = r0[0]; a0.q[1] = r0[1];
        a1.q[0] = r1[0]; a1.q[1] = r1[1];
      }
      if constexpr (BF16W) {
        const uint4* p = (const uint4*)(wot + (size_t)(h0 + m) * FDIM + k0 + 16 * lh);
        b.q[0] = p[0]; b.q[1] = p[1];
      } else {
        #pragma unroll
        for (int j = 0; j < 8; ++j) {
          int kk = k0 + 16 * lh + 2 * j;
          const float* p = wof + (size_t)kk * HDIM + h0 + m;
          b.u[j] = pack2(p[0], p[HDIM]);
        }
      }
      acc0 = __builtin_amdgcn_wmma_f32_16x16x32_bf16(false, a0.v, false, b.v, (short)0, acc0, false, false);
      acc1 = __builtin_amdgcn_wmma_f32_16x16x32_bf16(false, a1.v, false, b.v, (short)0, acc1, false, false);
    }
    #pragma unroll
    for (int r = 0; r < 8; ++r) {
      {
        int row = r + 8 * lh;
        float v = acc0[r] * sW[row];
        float* dst = out + (size_t)sTok[row] * HDIM + h0 + m;
        __hip_atomic_fetch_add(dst, v, __ATOMIC_RELAXED, __HIP_MEMORY_SCOPE_AGENT);
      }
      {
        int row = 16 + r + 8 * lh;
        float v = acc1[r] * sW[row];
        float* dst = out + (size_t)sTok[row] * HDIM + h0 + m;
        __hip_atomic_fetch_add(dst, v, __ATOMIC_RELAXED, __HIP_MEMORY_SCOPE_AGENT);
      }
    }
  }
}

extern "C" void kernel_launch(void* const* d_in, const int* in_sizes, int n_in,
                              void* d_out, int out_size, void* d_ws, size_t ws_size,
                              hipStream_t stream) {
  (void)in_sizes; (void)n_in; (void)out_size;
  const float* x   = (const float*)d_in[0];   // hidden_states [T,H]
  const float* tw  = (const float*)d_in[1];   // topk_weights  [T,K]
  const int*   ids = (const int*)  d_in[2];   // topk_ids      [T,K]
  const float* wi0 = (const float*)d_in[3];   // [E,H,F]
  const float* wi1 = (const float*)d_in[4];   // [E,H,F]
  const float* wo  = (const float*)d_in[5];   // [E,F,H]
  float* out = (float*)d_out;                 // [T,H]

  int* counts = (int*)d_ws;                              // 8 ints
  int* list   = (int*)((char*)d_ws + 256);               // E*TK ints (64 KB)
  const size_t tblOff  = 256 + (size_t)NEXP * TK * 4;    // 65792, 256B-aligned
  const size_t tblElems = (size_t)NEXP * MATELEMS;       // 8M bf16 per matrix set
  const size_t need = tblOff + 3 * tblElems * sizeof(unsigned short);  // ~48 MB

  zero_kernel<<<(T_TOK * HDIM / 4) / 256, 256, 0, stream>>>((float4*)out, counts);
  route_kernel<<<TK / 256, 256, 0, stream>>>(ids, counts, list);

  if (ws_size >= need) {
    unsigned short* t0 = (unsigned short*)((char*)d_ws + tblOff);
    unsigned short* t1 = t0 + tblElems;
    unsigned short* t2 = t1 + tblElems;
    wtrans_kernel<<<dim3(32, 32, 3 * NEXP), dim3(32, 8), 0, stream>>>(
        wi0, wi1, wo, t0, t1, t2);
    hipFuncSetAttribute((const void*)moe_kernel<true>,
                        hipFuncAttributeMaxDynamicSharedMemorySize, SMEM_BYTES);
    moe_kernel<true><<<NEXP * MAX_TILES, 256, SMEM_BYTES, stream>>>(
        x, tw, t0, t1, t2, counts, list, out);
  } else {
    hipFuncSetAttribute((const void*)moe_kernel<false>,
                        hipFuncAttributeMaxDynamicSharedMemorySize, SMEM_BYTES);
    moe_kernel<false><<<NEXP * MAX_TILES, 256, SMEM_BYTES, stream>>>(
        x, tw, wi0, wi1, wo, counts, list, out);
  }
}
